// GNNNet_86672440033767
// MI455X (gfx1250) — matrix operand order repaired and checked by reading, not compiled, not run on hardware
//
#include <hip/hip_runtime.h>
#include <math.h>

typedef __attribute__((ext_vector_type(16))) _Float16 v16h;
typedef __attribute__((ext_vector_type(8)))  _Float16 v8h;
typedef __attribute__((ext_vector_type(8)))  float    v8f;

#define N_NODES_C 102400
#define N_EDGES_C 1638400
#define NB_C      256
#define NPG_C     400
#define EN_LEN_C  420
#define EN_ROWS_C (NB_C * EN_LEN_C)
#define HSZ       26

// ---------------------------------------------------------------------------
// Utility kernels
// ---------------------------------------------------------------------------
__global__ void k_zero(float* __restrict__ p, long n) {
  long i = (long)blockIdx.x * blockDim.x + threadIdx.x;
  long stride = (long)gridDim.x * blockDim.x;
  for (; i < n; i += stride) p[i] = 0.f;
}

__global__ void k_deg(const int* __restrict__ ei, float* __restrict__ deg) {
  int e = blockIdx.x * blockDim.x + threadIdx.x;
  if (e < N_EDGES_C) atomicAdd(&deg[ei[N_EDGES_C + e]], 1.f);
}

__global__ void k_dinv(const float* __restrict__ deg, float* __restrict__ dinv) {
  int i = blockIdx.x * blockDim.x + threadIdx.x;
  if (i < N_NODES_C) dinv[i] = rsqrtf(deg[i] + 1.f);
}

// ---------------------------------------------------------------------------
// GEMM operand packing (fp32 -> zero-padded f16)
//   Ap[M, Kp] row-major  (M is always a multiple of 32 here)
//   Bt[Np, Kp] = B^T, with generic source strides B(k,n) = B[k*bs_k + n*bs_n]
// ---------------------------------------------------------------------------
__global__ void k_pack_a(const float* __restrict__ A, int lda,
                         _Float16* __restrict__ Ap, int Kp, int K, long total) {
  long idx = (long)blockIdx.x * blockDim.x + threadIdx.x;
  if (idx >= total) return;
  int k = (int)(idx % Kp);
  long row = idx / Kp;
  float v = (k < K) ? A[row * lda + k] : 0.f;
  Ap[idx] = (_Float16)v;
}

__global__ void k_pack_bt(const float* __restrict__ B, long bs_k, long bs_n,
                          _Float16* __restrict__ Bt, int Kp, int N, int K, long total) {
  long idx = (long)blockIdx.x * blockDim.x + threadIdx.x;
  if (idx >= total) return;
  int k = (int)(idx % Kp);
  int n = (int)(idx / Kp);
  float v = (n < N && k < K) ? B[(long)k * bs_k + (long)n * bs_n] : 0.f;
  Bt[idx] = (_Float16)v;
}

// ---------------------------------------------------------------------------
// WMMA GEMM on packed operands: C[M,N] = act(Ap @ Bt^T + bias)
// One wave per 32x32 output tile: 4 accumulators, 4 v_wmma per 32-wide K step.
// Fragments are contiguous vector loads straight from global (no LDS, no
// divergence around the WMMAs).
//   A frag layout: row = lane%16, K(e) = (lane&16?16:0) + e
//   B frag layout: col = lane%16, K(e) = (lane&16?16:0) + e
// ---------------------------------------------------------------------------
__launch_bounds__(32)
__global__ void k_gemm_wmma(const _Float16* __restrict__ Ap, int Kp,
                            const _Float16* __restrict__ Bt,
                            const float* __restrict__ bias,
                            float* __restrict__ C, int ldc,
                            int M, int N, int fuse_relu, int has_bias) {
  const int m0 = blockIdx.y * 32;
  const int n0 = blockIdx.x * 32;
  const int lane = threadIdx.x;          // wave32
  const int l15 = lane & 15;
  const int abase = (lane & 16) ? 16 : 0;  // A K-half select
  const int bbase = (lane & 16) ? 16 : 0;  // B K-half select

  const _Float16* arow0 = Ap + (long)(m0 + l15) * Kp;
  const _Float16* arow1 = Ap + (long)(m0 + 16 + l15) * Kp;
  const _Float16* brow0 = Bt + (long)(n0 + l15) * Kp;
  const _Float16* brow1 = Bt + (long)(n0 + 16 + l15) * Kp;

  v8f acc00 = {}, acc01 = {}, acc10 = {}, acc11 = {};

  for (int k0 = 0; k0 < Kp; k0 += 32) {
    union { v16h v; v8h h[2]; } a0, a1;
    a0.h[0] = *(const v8h*)(arow0 + k0 + abase);
    a0.h[1] = *(const v8h*)(arow0 + k0 + 8 + abase);
    a1.h[0] = *(const v8h*)(arow1 + k0 + abase);
    a1.h[1] = *(const v8h*)(arow1 + k0 + 8 + abase);
    v16h b0 = *(const v16h*)(brow0 + k0 + bbase);
    v16h b1 = *(const v16h*)(brow1 + k0 + bbase);

    acc00 = __builtin_amdgcn_wmma_f32_16x16x32_f16(false, a0.v, false, b0,
                                                   (short)0, acc00, false, false);
    acc01 = __builtin_amdgcn_wmma_f32_16x16x32_f16(false, a0.v, false, b1,
                                                   (short)0, acc01, false, false);
    acc10 = __builtin_amdgcn_wmma_f32_16x16x32_f16(false, a1.v, false, b0,
                                                   (short)0, acc10, false, false);
    acc11 = __builtin_amdgcn_wmma_f32_16x16x32_f16(false, a1.v, false, b1,
                                                   (short)0, acc11, false, false);
  }

  // Epilogue: C/D layout — VGPR r holds row (r + 8*(lane>=16)), col = lane%16.
  const int mb = m0 + ((lane & 16) ? 8 : 0);
  const int gn0 = n0 + l15;
  const int gn1 = n0 + 16 + l15;
  float bv0 = 0.f, bv1 = 0.f;
  if (has_bias) {
    if (gn0 < N) bv0 = bias[gn0];
    if (gn1 < N) bv1 = bias[gn1];
  }
#pragma unroll
  for (int r = 0; r < 8; ++r) {
    int gm0 = mb + r, gm1 = mb + 16 + r;
    if (gn0 < N) {
      if (gm0 < M) {
        float v = acc00[r] + bv0;
        if (fuse_relu) v = v > 0.f ? v : 0.f;
        C[(long)gm0 * ldc + gn0] = v;
      }
      if (gm1 < M) {
        float v = acc10[r] + bv0;
        if (fuse_relu) v = v > 0.f ? v : 0.f;
        C[(long)gm1 * ldc + gn0] = v;
      }
    }
    if (gn1 < N) {
      if (gm0 < M) {
        float v = acc01[r] + bv1;
        if (fuse_relu) v = v > 0.f ? v : 0.f;
        C[(long)gm0 * ldc + gn1] = v;
      }
      if (gm1 < M) {
        float v = acc11[r] + bv1;
        if (fuse_relu) v = v > 0.f ? v : 0.f;
        C[(long)gm1 * ldc + gn1] = v;
      }
    }
  }
}

// ---------------------------------------------------------------------------
// GCN edge aggregation: agg[dst,c] += h[src,c] * dinv[src]*dinv[dst]
// ---------------------------------------------------------------------------
__global__ void k_edge_agg(const float* __restrict__ h, const int* __restrict__ ei,
                           const float* __restrict__ dinv, float* __restrict__ agg,
                           int C) {
  long idx = (long)blockIdx.x * blockDim.x + threadIdx.x;
  long total = (long)N_EDGES_C * C;
  if (idx >= total) return;
  int e = (int)(idx / C);
  int c = (int)(idx % C);
  int s = ei[e];
  int d = ei[N_EDGES_C + e];
  float v = h[(long)s * C + c] * dinv[s] * dinv[d];
  atomicAdd(&agg[(long)d * C + c], v);
}

// out (in place over agg): relu(agg + h*dinv^2 + bias[c])
__global__ void k_gcn_post(const float* __restrict__ h, float* __restrict__ agg,
                           const float* __restrict__ dinv,
                           const float* __restrict__ bias, int C) {
  long idx = (long)blockIdx.x * blockDim.x + threadIdx.x;
  long total = (long)N_NODES_C * C;
  if (idx >= total) return;
  int node = (int)(idx / C);
  int c = (int)(idx % C);
  float di = dinv[node];
  float v = agg[idx] + h[idx] * di * di + bias[c];
  agg[idx] = v > 0.f ? v : 0.f;
}

// Mean-pool over contiguous groups of L rows: out[g,c] = mean_i in[g*L+i, c]
__global__ void k_pool(const float* __restrict__ in, float* __restrict__ out,
                       int groups, int L, int C) {
  int idx = blockIdx.x * blockDim.x + threadIdx.x;
  if (idx >= groups * C) return;
  int g = idx / C, c = idx % C;
  const float* base = in + (long)g * L * C + c;
  float s = 0.f;
  for (int i = 0; i < L; ++i) s += base[(long)i * C];
  out[(long)g * C + c] = s * (1.f / (float)L);
}

// Eval-mode BN (running mean 0 / var 1): out = in * s * g[c] + b[c]
__global__ void k_bn(const float* __restrict__ in, float* __restrict__ out,
                     const float* __restrict__ g, const float* __restrict__ b,
                     long rows, int C) {
  long idx = (long)blockIdx.x * blockDim.x + threadIdx.x;
  if (idx >= rows * C) return;
  int c = (int)(idx % C);
  const float s = rsqrtf(1.f + 1e-5f);
  out[idx] = in[idx] * s * g[c] + b[c];
}

__global__ void k_concat(const float* __restrict__ a, const float* __restrict__ b,
                         float* __restrict__ out) {
  int idx = blockIdx.x * blockDim.x + threadIdx.x;   // NB_C * 256
  if (idx >= NB_C * 256) return;
  int r = idx >> 8, c = idx & 255;
  out[idx] = (c < 128) ? a[r * 128 + c] : b[r * 128 + (c - 128)];
}

// One LSTM cell step with zero (h,c): h = sig(z_o) * tanh(sig(z_i)*tanh(z_g))
// Wih row-major [4H, K]; gates order i,f,g,o; f unused (c_prev==0).
__global__ void k_lstm(const float* __restrict__ xin, int K,
                       const float* __restrict__ Wih, const float* __restrict__ bvec,
                       float* __restrict__ hout, int ldh, int hoff) {
  int idx = blockIdx.x * blockDim.x + threadIdx.x;
  if (idx >= NB_C * HSZ) return;
  int r = idx / HSZ, u = idx % HSZ;
  const float* xr = xin + (long)r * K;
  float zi = bvec[u];
  float zg = bvec[2 * HSZ + u];
  float zo = bvec[3 * HSZ + u];
  const float* wi = Wih + (long)u * K;
  const float* wg = Wih + (long)(2 * HSZ + u) * K;
  const float* wo = Wih + (long)(3 * HSZ + u) * K;
  for (int k = 0; k < K; ++k) {
    float xv = xr[k];
    zi += xv * wi[k];
    zg += xv * wg[k];
    zo += xv * wo[k];
  }
  float si = 1.f / (1.f + expf(-zi));
  float so = 1.f / (1.f + expf(-zo));
  float cc = si * tanhf(zg);
  hout[(long)r * ldh + hoff + u] = so * tanhf(cc);
}

// Final head: out[r] = sigmoid(z[r,:] . Wout + bout)
__global__ void k_final(const float* __restrict__ z, const float* __restrict__ Wout,
                        const float* __restrict__ bout, float* __restrict__ out) {
  int r = blockIdx.x * blockDim.x + threadIdx.x;
  if (r >= NB_C) return;
  float acc = bout[0];
  const float* zr = z + (long)r * 128;
  for (int k = 0; k < 128; ++k) acc += zr[k] * Wout[k];
  out[r] = 1.f / (1.f + expf(-acc));
}

// ---------------------------------------------------------------------------
// Host-side orchestration
// ---------------------------------------------------------------------------
static inline unsigned blks(long n, int t) { return (unsigned)((n + t - 1) / t); }

static void run_gemm(const float* A, int lda, const float* B, long bs_k, long bs_n,
                     const float* bias, float* C, int ldc, int M, int N, int K,
                     int relu, int has_bias, _Float16* Ap, _Float16* Bt,
                     hipStream_t stream) {
  int Kp = ((K + 31) / 32) * 32;
  int Np = ((N + 31) / 32) * 32;
  long na = (long)M * Kp;
  long nb = (long)Np * Kp;
  k_pack_a<<<blks(na, 256), 256, 0, stream>>>(A, lda, Ap, Kp, K, na);
  k_pack_bt<<<blks(nb, 256), 256, 0, stream>>>(B, bs_k, bs_n, Bt, Kp, N, K, nb);
  dim3 g((unsigned)(Np / 32), (unsigned)(M / 32));   // M always multiple of 32
  k_gemm_wmma<<<g, 32, 0, stream>>>(Ap, Kp, Bt, bias, C, ldc, M, N, relu, has_bias);
}

extern "C" void kernel_launch(void* const* d_in, const int* in_sizes, int n_in,
                              void* d_out, int out_size, void* d_ws, size_t ws_size,
                              hipStream_t stream) {
  (void)in_sizes; (void)n_in; (void)out_size; (void)ws_size;

  const float* x     = (const float*)d_in[0];
  const int*   ei    = (const int*)  d_in[1];
  const float* en    = (const float*)d_in[3];
  const float* W1    = (const float*)d_in[4];
  const float* b1    = (const float*)d_in[5];
  const float* W2    = (const float*)d_in[6];
  const float* b2    = (const float*)d_in[7];
  const float* Wg    = (const float*)d_in[8];
  const float* bg    = (const float*)d_in[9];
  const float* bn0g  = (const float*)d_in[10];
  const float* bn0b  = (const float*)d_in[11];
  const float* Wc1   = (const float*)d_in[12];
  const float* bc1   = (const float*)d_in[13];
  const float* bn1g  = (const float*)d_in[14];
  const float* bn1b  = (const float*)d_in[15];
  const float* Wc2   = (const float*)d_in[16];
  const float* bc2   = (const float*)d_in[17];
  const float* bn2g  = (const float*)d_in[18];
  const float* bn2b  = (const float*)d_in[19];
  const float* We    = (const float*)d_in[20];
  const float* be    = (const float*)d_in[21];
  const float* Wfc1  = (const float*)d_in[22];
  const float* bfc1  = (const float*)d_in[23];
  const float* Wih0  = (const float*)d_in[24];
  const float* Wih12 = (const float*)d_in[25];
  const float* lstmb = (const float*)d_in[26];
  const float* Wfc2  = (const float*)d_in[27];
  const float* bfc2  = (const float*)d_in[28];
  const float* Wout  = (const float*)d_in[29];
  const float* bout  = (const float*)d_in[30];
  float* out = (float*)d_out;

  // Workspace layout (floats). Regions R2..R4 are reused between branches.
  float* ws = (float*)d_ws;
  float* deg   = ws;                       // 102400
  float* dinv  = deg + N_NODES_C;          // 102400
  float* R2    = dinv + N_NODES_C;         // 5,529,600  (h1 | bn0)
  float* R3    = R2 + 5529600;             // 10,752,000 (agg1->xt1 | c1)
  float* R4    = R3 + 10752000;            // 11,059,200 (h2 | c2)
  float* R5    = R4 + 11059200;            // 11,059,200 (agg2->xt2)
  float* sm    = R5 + 11059200;
  float* pool_g  = sm;                     // 256*108
  float* xtg     = pool_g + NB_C * 112;    // 256*128 (offset padded to 16)
  float* pool_en = xtg + NB_C * 128;       // 256*100
  float* h_en    = pool_en + NB_C * 112;   // 256*128
  float* xc      = h_en + NB_C * 128;      // 256*256
  float* fc1o    = xc + NB_C * 256;        // 256*128
  float* hA      = fc1o + NB_C * 128;      // 256*64 (52 used)
  float* hB      = hA + NB_C * 64;         // 256*64
  float* zbuf    = hB + NB_C * 64;         // 256*128
  float* packs   = zbuf + NB_C * 128;
  _Float16* Ap = (_Float16*)packs;         // up to 107520*128 halfs = 27.5 MB
  _Float16* Bt = (_Float16*)(packs + 6881280);  // up to 128*256 halfs

  const int T = 256;

  // ---- degree / normalization ----
  k_zero<<<blks(N_NODES_C, T), T, 0, stream>>>(deg, N_NODES_C);
  k_deg<<<blks(N_EDGES_C, T), T, 0, stream>>>(ei, deg);
  k_dinv<<<blks(N_NODES_C, T), T, 0, stream>>>(deg, dinv);

  // ---- GCN layer 1: h1 = x @ W1 ----
  run_gemm(x, 54, W1, 54, 1, nullptr, R2, 54, N_NODES_C, 54, 54, 0, 0, Ap, Bt, stream);
  k_zero<<<blks((long)N_NODES_C * 54, T), T, 0, stream>>>(R3, (long)N_NODES_C * 54);
  k_edge_agg<<<blks((long)N_EDGES_C * 54, T), T, 0, stream>>>(R2, ei, dinv, R3, 54);
  k_gcn_post<<<blks((long)N_NODES_C * 54, T), T, 0, stream>>>(R2, R3, dinv, b1, 54);

  // ---- GCN layer 2: h2 = xt1 @ W2 ----
  run_gemm(R3, 54, W2, 108, 1, nullptr, R4, 108, N_NODES_C, 108, 54, 0, 0, Ap, Bt, stream);
  k_zero<<<blks((long)N_NODES_C * 108, T), T, 0, stream>>>(R5, (long)N_NODES_C * 108);
  k_edge_agg<<<blks((long)N_EDGES_C * 108, T), T, 0, stream>>>(R4, ei, dinv, R5, 108);
  k_gcn_post<<<blks((long)N_NODES_C * 108, T), T, 0, stream>>>(R4, R5, dinv, b2, 108);

  // ---- global mean pool + Wg FC ----
  k_pool<<<blks(NB_C * 108, T), T, 0, stream>>>(R5, pool_g, NB_C, NPG_C, 108);
  run_gemm(pool_g, 108, Wg, 128, 1, bg, xtg, 128, NB_C, 128, 108, 1, 1, Ap, Bt, stream);

  // ---- en branch ----
  k_bn<<<blks((long)EN_ROWS_C * 34, T), T, 0, stream>>>(en, R2, bn0g, bn0b, EN_ROWS_C, 34);
  // conv1 (center tap): B(k=in,n=out) = Wc1[n*102 + k*3 + 1]
  run_gemm(R2, 34, Wc1 + 1, 3, 102, bc1, R3, 100, EN_ROWS_C, 100, 34, 1, 1, Ap, Bt, stream);
  k_bn<<<blks((long)EN_ROWS_C * 100, T), T, 0, stream>>>(R3, R3, bn1g, bn1b, EN_ROWS_C, 100);
  // conv2 (center tap): B(k,n) = Wc2[n*300 + k*3 + 1]
  run_gemm(R3, 100, Wc2 + 1, 3, 300, bc2, R4, 100, EN_ROWS_C, 100, 100, 1, 1, Ap, Bt, stream);
  k_bn<<<blks((long)EN_ROWS_C * 100, T), T, 0, stream>>>(R4, R4, bn2g, bn2b, EN_ROWS_C, 100);
  k_pool<<<blks(NB_C * 100, T), T, 0, stream>>>(R4, pool_en, NB_C, EN_LEN_C, 100);
  run_gemm(pool_en, 100, We, 128, 1, be, h_en, 128, NB_C, 128, 100, 1, 1, Ap, Bt, stream);

  // ---- fuse + fc1 ----
  k_concat<<<blks(NB_C * 256, T), T, 0, stream>>>(h_en, xtg, xc);
  run_gemm(xc, 256, Wfc1, 128, 1, bfc1, fc1o, 128, NB_C, 128, 256, 1, 1, Ap, Bt, stream);

  // ---- 3-layer bidirectional LSTM (length-1 sequences, zero init) ----
  const int L0 = 4 * HSZ * 128;   // per-direction Wih0 size
  const int L12 = 4 * HSZ * 52;   // per-(layer,dir) Wih12 size
  // layer 0
  k_lstm<<<blks(NB_C * HSZ, T), T, 0, stream>>>(fc1o, 128, Wih0,      lstmb,           hA, 52, 0);
  k_lstm<<<blks(NB_C * HSZ, T), T, 0, stream>>>(fc1o, 128, Wih0 + L0, lstmb + 104,     hA, 52, HSZ);
  // layer 1
  k_lstm<<<blks(NB_C * HSZ, T), T, 0, stream>>>(hA, 52, Wih12,           lstmb + 2 * 104, hB, 52, 0);
  k_lstm<<<blks(NB_C * HSZ, T), T, 0, stream>>>(hA, 52, Wih12 + L12,     lstmb + 3 * 104, hB, 52, HSZ);
  // layer 2
  k_lstm<<<blks(NB_C * HSZ, T), T, 0, stream>>>(hB, 52, Wih12 + 2 * L12, lstmb + 4 * 104, hA, 52, 0);
  k_lstm<<<blks(NB_C * HSZ, T), T, 0, stream>>>(hB, 52, Wih12 + 3 * L12, lstmb + 5 * 104, hA, 52, HSZ);

  // ---- fc2 + sigmoid head ----
  run_gemm(hA, 52, Wfc2, 128, 1, bfc2, zbuf, 128, NB_C, 128, 52, 0, 1, Ap, Bt, stream);
  k_final<<<blks(NB_C, T), T, 0, stream>>>(zbuf, Wout, bout, out);
}
